// VisionMHAttention_76467597738333
// MI455X (gfx1250) — matrix operand order repaired and checked by reading, not compiled, not run on hardware
//
#include <hip/hip_runtime.h>

// ---------------------------------------------------------------------------
// VisionMHAttention for MI455X (gfx1250): bf16 WMMA + async global->LDS DMA.
//   B=8, N=1024, C=768, H=12, D=64, scale = 1/8
// ---------------------------------------------------------------------------

#define BATCH   8
#define SEQ     1024
#define EMBED   768
#define HEADS   12
#define HDIM    64
#define ROWS    (BATCH * SEQ)          // 8192
#define QKVN    (3 * EMBED)            // 2304
#define SCALER  0.125f

typedef __attribute__((ext_vector_type(16))) __bf16 v16bf;
typedef __attribute__((ext_vector_type(8)))  float  v8f;

union Frag {
    v16bf v;
    uint4 u[2];
};

__device__ __forceinline__ unsigned short f2bf(float f) {
    unsigned int u = __float_as_uint(f);
    u += 0x7FFFu + ((u >> 16) & 1u);          // round-to-nearest-even
    return (unsigned short)(u >> 16);
}

// Low 32 bits of a flat LDS pointer = workgroup-relative LDS byte offset.
__device__ __forceinline__ unsigned lds_off(const void* p) {
    return (unsigned)(size_t)p;
}

// CDNA5 async DMA: global -> LDS, 16B per lane, tracked by ASYNCcnt.
__device__ __forceinline__ void async_ld128(unsigned lds_byte_off, const void* gsrc) {
    asm volatile("global_load_async_to_lds_b128 %0, %1, off"
                 :: "v"(lds_byte_off), "v"((unsigned long long)(size_t)gsrc)
                 : "memory");
}
#define WAIT_ASYNC() asm volatile("s_wait_asynccnt 0x0" ::: "memory")

// ---------------------------------------------------------------------------
// fp32 -> bf16 converts
// ---------------------------------------------------------------------------
__global__ __launch_bounds__(256) void k_cvt(const float* __restrict__ in,
                                             unsigned short* __restrict__ out, int n) {
    int i = blockIdx.x * 256 + threadIdx.x;
    if (i < n) out[i] = f2bf(in[i]);
}

// in [rows][cols] fp32 -> out [cols][rows] bf16, LDS-tiled 32x32 (coalesced both ways)
__global__ __launch_bounds__(256) void k_cvt_t(const float* __restrict__ in,
                                               unsigned short* __restrict__ out,
                                               int rows, int cols) {
    __shared__ unsigned short t[32][33];
    const int tx = threadIdx.x & 31, ty = threadIdx.x >> 5;   // 32 x 8
    const int tr = blockIdx.y * 32, tc = blockIdx.x * 32;
#pragma unroll
    for (int i = 0; i < 4; ++i) {
        int r = tr + ty + i * 8;
        t[ty + i * 8][tx] = f2bf(in[(size_t)r * cols + tc + tx]);
    }
    __syncthreads();
#pragma unroll
    for (int i = 0; i < 4; ++i) {
        int r = tc + ty + i * 8;                              // out [cols][rows]
        out[(size_t)r * rows + tr + tx] = t[tx][ty + i * 8];
    }
}

// ---------------------------------------------------------------------------
// 128x128 WMMA GEMM core, double-buffered async-DMA staging.
// BT[N][K] bf16 (contiguous K). 256 threads = 8 waves; wave owns 32x64 =
// 2x4 WMMA tiles. LDS stride 56 halfwords (28 dwords; 28*l mod 64 distinct
// for l=0..15 -> conflict-free fragment reads).
// ---------------------------------------------------------------------------
#define LDT      56
#define TILE_HW  (128 * LDT)           // halfwords per buffer
#define TILE_B   (TILE_HW * 2)         // bytes per buffer

__device__ __forceinline__ void gemm128x128(const unsigned short* __restrict__ A,
                                            const unsigned short* __restrict__ BT,
                                            int K, int m_base, int n_base,
                                            unsigned short* lA, unsigned short* lB,
                                            v8f acc[2][4]) {
    const int tid  = threadIdx.x;
    const int lane = tid & 31, wave = tid >> 5;
    const int lh = lane & 15, hi = lane >> 4;
    const int wm = (wave >> 1) << 5;          // 0,32,64,96
    const int wn = (wave & 1) << 6;           // 0,64
    const int lrow = tid >> 1;                // 0..127
    const int lk   = (tid & 1) << 4;          // 0 or 16
    const v8f vzero = {0.f,0.f,0.f,0.f,0.f,0.f,0.f,0.f};
#pragma unroll
    for (int i = 0; i < 2; ++i)
#pragma unroll
        for (int j = 0; j < 4; ++j) acc[i][j] = vzero;

    const unsigned short* gA = A  + (size_t)(m_base + lrow) * K + lk;
    const unsigned short* gB = BT + (size_t)(n_base + lrow) * K + lk;
    const unsigned laOff = lds_off(lA + lrow * LDT + lk);     // bytes
    const unsigned lbOff = lds_off(lB + lrow * LDT + lk);

    // prologue: DMA tile 0 into buffer 0
    async_ld128(laOff,      gA);
    async_ld128(laOff + 16, gA + 8);
    async_ld128(lbOff,      gB);
    async_ld128(lbOff + 16, gB + 8);

    for (int k0 = 0; k0 < K; k0 += 32) {
        const int cur = (k0 >> 5) & 1;
        WAIT_ASYNC();                         // my tile-k DMAs landed
        __syncthreads();                      // everyone's landed; prev reads done
        if (k0 + 32 < K) {                    // DMA tile k+1 into other buffer
            const int nxt = cur ^ 1;
            async_ld128(laOff + nxt * TILE_B,      gA + k0 + 32);
            async_ld128(laOff + nxt * TILE_B + 16, gA + k0 + 40);
            async_ld128(lbOff + nxt * TILE_B,      gB + k0 + 32);
            async_ld128(lbOff + nxt * TILE_B + 16, gB + k0 + 40);
        }
        const unsigned short* cA = lA + cur * TILE_HW;
        const unsigned short* cB = lB + cur * TILE_HW;

        Frag af[2];
#pragma unroll
        for (int i = 0; i < 2; ++i) {
            const unsigned short* p = cA + (wm + i * 16 + lh) * LDT + hi * 8;
            af[i].u[0] = *(const uint4*)p;        // K 0..7   (hi: 8..15)
            af[i].u[1] = *(const uint4*)(p + 16); // K 16..23 (hi: 24..31)
        }
#pragma unroll
        for (int j = 0; j < 4; ++j) {
            Frag bf;
            const unsigned short* p = cB + (wn + j * 16 + lh) * LDT + hi * 8;
            bf.u[0] = *(const uint4*)p;           // K 0..7   (hi: 8..15)
            bf.u[1] = *(const uint4*)(p + 16);    // K 16..23 (hi: 24..31)
#pragma unroll
            for (int i = 0; i < 2; ++i)
                acc[i][j] = __builtin_amdgcn_wmma_f32_16x16x32_bf16(
                    false, af[i].v, false, bf.v, (short)0, acc[i][j], false, false);
        }
    }
}

// ---------------------------------------------------------------------------
// QKV GEMM + scatter:  q,k -> [B][H][N][D] bf16 (q pre-scaled), v -> [B][H][D][N]
// ---------------------------------------------------------------------------
__global__ __launch_bounds__(256) void k_gemm_qkv(const unsigned short* __restrict__ A,
                                                  const unsigned short* __restrict__ BT,
                                                  const float* __restrict__ bias,
                                                  unsigned short* __restrict__ qb,
                                                  unsigned short* __restrict__ kb,
                                                  unsigned short* __restrict__ vT) {
    __shared__ __align__(16) unsigned short lA[2 * TILE_HW];
    __shared__ __align__(16) unsigned short lB[2 * TILE_HW];
    v8f acc[2][4];
    const int m_base = blockIdx.y * 128, n_base = blockIdx.x * 128;
    gemm128x128(A, BT, EMBED, m_base, n_base, lA, lB, acc);

    const int lane = threadIdx.x & 31, wave = threadIdx.x >> 5;
    const int lh = lane & 15, hi = lane >> 4;
    const int wm = (wave >> 1) << 5, wn = (wave & 1) << 6;
#pragma unroll
    for (int i = 0; i < 2; ++i) {
#pragma unroll
        for (int j = 0; j < 4; ++j) {
            int c = n_base + wn + j * 16 + lh;          // 0..2303
            int part = c / EMBED;                       // 0=q 1=k 2=v
            int rem = c - part * EMBED;
            int h = rem >> 6, d = rem & 63;
            float bv = bias[c];
#pragma unroll
            for (int r = 0; r < 8; ++r) {
                int m = m_base + wm + i * 16 + hi * 8 + r;
                int b = m >> 10, nseq = m & 1023;
                float val = acc[i][j][r] + bv;
                size_t bh = (size_t)(b * HEADS + h);
                if (part == 0)      qb[(bh * SEQ + nseq) * HDIM + d] = f2bf(val * SCALER);
                else if (part == 1) kb[(bh * SEQ + nseq) * HDIM + d] = f2bf(val);
                else                vT[(bh * HDIM + d) * SEQ + nseq] = f2bf(val);
            }
        }
    }
}

// ---------------------------------------------------------------------------
// Output projection GEMM: fp32 + bias
// ---------------------------------------------------------------------------
__global__ __launch_bounds__(256) void k_gemm_out(const unsigned short* __restrict__ A,
                                                  const unsigned short* __restrict__ BT,
                                                  const float* __restrict__ bias,
                                                  float* __restrict__ C) {
    __shared__ __align__(16) unsigned short lA[2 * TILE_HW];
    __shared__ __align__(16) unsigned short lB[2 * TILE_HW];
    v8f acc[2][4];
    const int m_base = blockIdx.y * 128, n_base = blockIdx.x * 128;
    gemm128x128(A, BT, EMBED, m_base, n_base, lA, lB, acc);

    const int lane = threadIdx.x & 31, wave = threadIdx.x >> 5;
    const int lh = lane & 15, hi = lane >> 4;
    const int wm = (wave >> 1) << 5, wn = (wave & 1) << 6;
#pragma unroll
    for (int i = 0; i < 2; ++i) {
#pragma unroll
        for (int j = 0; j < 4; ++j) {
            int col = n_base + wn + j * 16 + lh;
            float bv = bias[col];
#pragma unroll
            for (int r = 0; r < 8; ++r) {
                int row = m_base + wm + i * 16 + hi * 8 + r;
                C[(size_t)row * EMBED + col] = acc[i][j][r] + bv;
            }
        }
    }
}

// ---------------------------------------------------------------------------
// Flash attention: block = 8 waves x 16 query rows = 128 q-rows of one (b,h).
// 32-key K [32][64] and V^T [64][32] tiles double-buffered via async DMA,
// online softmax in registers, P relayout via per-wave LDS (in-order DS).
// ---------------------------------------------------------------------------
#define LDK   72   // 36 dwords: 36*l mod 64 distinct -> conflict-free
#define LDV   56
#define LDP   56
#define KT_HW (32 * LDK)
#define KT_B  (KT_HW * 2)
#define VT_HW (64 * LDV)
#define VT_B  (VT_HW * 2)

__global__ __launch_bounds__(256) void k_attn(const unsigned short* __restrict__ q,
                                              const unsigned short* __restrict__ k,
                                              const unsigned short* __restrict__ vT,
                                              unsigned short* __restrict__ o) {
    __shared__ __align__(16) unsigned short lK[2 * KT_HW];
    __shared__ __align__(16) unsigned short lV[2 * VT_HW];
    __shared__ __align__(16) unsigned short lP[8][16 * LDP];

    const int tid  = threadIdx.x;
    const int lane = tid & 31, wave = tid >> 5;
    const int lh = lane & 15, hi = lane >> 4;
    const int bh = blockIdx.x >> 3;           // 0..95  (b*12+h)
    const int qt = blockIdx.x & 7;
    const int qrow = qt * 128 + wave * 16;

    const unsigned short* qp = q  + ((size_t)bh * SEQ + qrow) * HDIM;
    const unsigned short* kp = k  + (size_t)bh * SEQ * HDIM;
    const unsigned short* vp = vT + (size_t)bh * HDIM * SEQ;

    // Q fragments (16 rows x 64 d = two 16x32 A-fragments), Q pre-scaled
    Frag aq[2];
#pragma unroll
    for (int s = 0; s < 2; ++s) {
        const unsigned short* p = qp + lh * HDIM + s * 32 + hi * 8;
        aq[s].u[0] = *(const uint4*)p;
        aq[s].u[1] = *(const uint4*)(p + 16);
    }

    const v8f vzero = {0.f,0.f,0.f,0.f,0.f,0.f,0.f,0.f};
    v8f O[4] = {vzero, vzero, vzero, vzero};
    float m_[8], l_[8];
#pragma unroll
    for (int r = 0; r < 8; ++r) { m_[r] = -1e30f; l_[r] = 0.f; }

    // per-thread DMA slots: K tile 32 keys x 64 d; V tile 64 d x 32 keys
    const int kkey = tid >> 3, kseg = tid & 7;
    const int vd   = tid >> 2, vseg = tid & 3;
    const unsigned short* gK = kp + (size_t)kkey * HDIM + kseg * 8;
    const unsigned short* gV = vp + (size_t)vd * SEQ + vseg * 8;
    const unsigned lkOff = lds_off(lK + kkey * LDK + kseg * 8);
    const unsigned lvOff = lds_off(lV + vd * LDV + vseg * 8);

    // prologue: DMA key-tile 0 into buffer 0
    async_ld128(lkOff, gK);
    async_ld128(lvOff, gV);

    for (int kt = 0; kt < SEQ; kt += 32) {
        const int cur = (kt >> 5) & 1;
        WAIT_ASYNC();
        __syncthreads();
        if (kt + 32 < SEQ) {
            const int nxt = cur ^ 1;
            async_ld128(lkOff + nxt * KT_B, gK + (size_t)(kt + 32) * HDIM);
            async_ld128(lvOff + nxt * VT_B, gV + kt + 32);
        }
        const unsigned short* cK = lK + cur * KT_HW;
        const unsigned short* cV = lV + cur * VT_HW;

        // S = Q * K^T  (16 q x 32 keys), two N-subtiles, two d-steps each
        v8f S[2] = {vzero, vzero};
#pragma unroll
        for (int t = 0; t < 2; ++t) {
#pragma unroll
            for (int s = 0; s < 2; ++s) {
                Frag bk;
                const unsigned short* p = cK + (t * 16 + lh) * LDK + s * 32 + hi * 8;
                bk.u[0] = *(const uint4*)p;
                bk.u[1] = *(const uint4*)(p + 16);
                S[t] = __builtin_amdgcn_wmma_f32_16x16x32_bf16(
                    false, aq[s].v, false, bk.v, (short)0, S[t], false, false);
            }
        }

        // online softmax (row stats; 16-lane row reductions stay in-half)
#pragma unroll
        for (int r = 0; r < 8; ++r) {
            float s0 = S[0][r], s1 = S[1][r];
            float mr = fmaxf(s0, s1);
            mr = fmaxf(mr, __shfl_xor(mr, 1, 32));
            mr = fmaxf(mr, __shfl_xor(mr, 2, 32));
            mr = fmaxf(mr, __shfl_xor(mr, 4, 32));
            mr = fmaxf(mr, __shfl_xor(mr, 8, 32));
            float mn = fmaxf(m_[r], mr);
            float sc = __expf(m_[r] - mn);
            m_[r] = mn;
            float p0 = __expf(s0 - mn), p1 = __expf(s1 - mn);
            S[0][r] = p0; S[1][r] = p1;
            float rs = p0 + p1;
            rs += __shfl_xor(rs, 1, 32);
            rs += __shfl_xor(rs, 2, 32);
            rs += __shfl_xor(rs, 4, 32);
            rs += __shfl_xor(rs, 8, 32);
            l_[r] = l_[r] * sc + rs;
#pragma unroll
            for (int j = 0; j < 4; ++j) O[j][r] = O[j][r] * sc;
        }

        // P (C-layout) -> per-wave LDS -> A-layout fragment
        unsigned short* pw = &lP[wave][0];
#pragma unroll
        for (int r = 0; r < 8; ++r) {
            int row = hi * 8 + r;
            pw[row * LDP + lh]      = f2bf(S[0][r]);
            pw[row * LDP + 16 + lh] = f2bf(S[1][r]);
        }
        asm volatile("s_wait_dscnt 0x0" ::: "memory");  // wave-local DS handoff

        Frag ap;
        {
            const unsigned short* p = pw + lh * LDP + hi * 8;
            ap.u[0] = *(const uint4*)p;
            ap.u[1] = *(const uint4*)(p + 16);
        }
#pragma unroll
        for (int j = 0; j < 4; ++j) {
            Frag bv;
            const unsigned short* p = cV + (j * 16 + lh) * LDV + hi * 8;
            bv.u[0] = *(const uint4*)p;
            bv.u[1] = *(const uint4*)(p + 16);
            O[j] = __builtin_amdgcn_wmma_f32_16x16x32_bf16(
                false, ap.v, false, bv.v, (short)0, O[j], false, false);
        }
    }

    // normalize + store to attn_out [ROWS][EMBED] bf16
    const int b = bh / HEADS, h = bh - b * HEADS;
#pragma unroll
    for (int r = 0; r < 8; ++r) {
        float inv = 1.0f / l_[r];
        int row = b * SEQ + qrow + hi * 8 + r;
#pragma unroll
        for (int j = 0; j < 4; ++j) {
            int col = h * HDIM + j * 16 + lh;
            o[(size_t)row * EMBED + col] = f2bf(O[j][r] * inv);
        }
    }
}

// ---------------------------------------------------------------------------
// Host-side orchestration
// ---------------------------------------------------------------------------
extern "C" void kernel_launch(void* const* d_in, const int* in_sizes, int n_in,
                              void* d_out, int out_size, void* d_ws, size_t ws_size,
                              hipStream_t stream) {
    const float* x      = (const float*)d_in[0];
    const float* w_qkv  = (const float*)d_in[1];
    const float* b_qkv  = (const float*)d_in[2];
    const float* w_out  = (const float*)d_in[3];
    const float* b_out  = (const float*)d_in[4];
    float* out = (float*)d_out;

    unsigned short* ws = (unsigned short*)d_ws;
    unsigned short* x_bf   = ws;                               // 8192*768
    unsigned short* wqkvT  = x_bf  + (size_t)ROWS * EMBED;     // 2304*768
    unsigned short* woutT  = wqkvT + (size_t)QKVN * EMBED;     // 768*768
    unsigned short* qbuf   = woutT + (size_t)EMBED * EMBED;    // 8*12*1024*64
    unsigned short* kbuf   = qbuf  + (size_t)ROWS * EMBED;
    unsigned short* vTbuf  = kbuf  + (size_t)ROWS * EMBED;
    unsigned short* attn_o = vTbuf + (size_t)ROWS * EMBED;

    // 1) precision converts
    {
        int n = ROWS * EMBED;
        k_cvt<<<(n + 255) / 256, 256, 0, stream>>>(x, x_bf, n);
    }
    k_cvt_t<<<dim3(QKVN / 32, EMBED / 32), 256, 0, stream>>>(w_qkv, wqkvT, EMBED, QKVN);
    k_cvt_t<<<dim3(EMBED / 32, EMBED / 32), 256, 0, stream>>>(w_out, woutT, EMBED, EMBED);

    // 2) QKV GEMM + scatter  (M=8192, N=2304, K=768)
    k_gemm_qkv<<<dim3(QKVN / 128, ROWS / 128), 256, 0, stream>>>(
        x_bf, wqkvT, b_qkv, qbuf, kbuf, vTbuf);

    // 3) flash attention  (96 head-batches x 8 query tiles of 128 rows)
    k_attn<<<BATCH * HEADS * (SEQ / 128), 256, 0, stream>>>(qbuf, kbuf, vTbuf, attn_o);

    // 4) output projection  (M=8192, N=768, K=768)
    k_gemm_out<<<dim3(EMBED / 128, ROWS / 128), 256, 0, stream>>>(
        attn_o, woutT, b_out, out);
}
